// Visnet_33741263077377
// MI455X (gfx1250) — matrix operand order repaired and checked by reading, not compiled, not run on hardware
//
#include <hip/hip_runtime.h>
#include <hip/hip_bf16.h>
#include <cstddef>

#define NNODES 10000
#define NEDGES 160000
#define H 128
#define LDIM 8

typedef __bf16 bf16_t;
typedef __bf16 v16bf __attribute__((ext_vector_type(16)));
typedef __bf16 bfx8  __attribute__((ext_vector_type(8)));
typedef float  v8f   __attribute__((ext_vector_type(8)));

// ---------------------------------------------------------------------------
// helpers
// ---------------------------------------------------------------------------
__device__ __forceinline__ float silu_f(float x) {
    // native v_rcp_f32 path — downstream consumers are bf16/atomic adds,
    // IEEE-division precision is wasted here.
    return x * __builtin_amdgcn_rcpf(1.0f + __expf(-x));
}

__device__ __forceinline__ float wave_sum32(float v) {
    v += __shfl_xor(v, 1);
    v += __shfl_xor(v, 2);
    v += __shfl_xor(v, 4);
    v += __shfl_xor(v, 8);
    v += __shfl_xor(v, 16);
    return v;
}

__device__ __forceinline__ void atomic_add_f32(float* p, float v) {
    __hip_atomic_fetch_add(p, v, __ATOMIC_RELAXED, __HIP_MEMORY_SCOPE_AGENT);
}

// ---------------------------------------------------------------------------
// fp32 -> bf16 convert
// ---------------------------------------------------------------------------
__global__ void f32_to_bf16_kernel(const float* __restrict__ in,
                                   bf16_t* __restrict__ out, unsigned n) {
    unsigned i = blockIdx.x * blockDim.x + threadIdx.x;
    if (i < n) out[i] = (bf16_t)in[i];
}

// ---------------------------------------------------------------------------
// Pack fp32 weight W[128, Nc] (row-major) into WMMA B-fragment order (bf16):
// P[((nt*4 + kc)*32 + lane)*16 + h] = W[kc*32 + (lane>>4)*16 + h, nt*16 + (lane&15)]
// (16-bit B 32x16, wave32, contiguous k order also used for A — lanes 0-15 K=0..15, 16-31 K=16..31)
// ---------------------------------------------------------------------------
__global__ void pack_weight_kernel(const float* __restrict__ W,
                                   bf16_t* __restrict__ P, int Nc) {
    int i = blockIdx.x * blockDim.x + threadIdx.x;
    if (i >= 128 * Nc) return;
    int h    = i & 15;
    int lane = (i >> 4) & 31;
    int kc   = (i >> 9) & 3;
    int nt   = i >> 11;
    int K    = kc * 32 + (lane >> 4) * 16 + h;
    int col  = nt * 16 + (lane & 15);
    P[i] = (bf16_t)W[(unsigned)(K * Nc + col)];
}

// ---------------------------------------------------------------------------
// Node prep: LayerNorm(node_feats) -> x_bf16 ; vec = vector_feats*vln_w
// ---------------------------------------------------------------------------
__global__ void node_prep_kernel(const float* __restrict__ node_feats,
                                 const float* __restrict__ vector_feats,
                                 const float* __restrict__ ln_scale,
                                 const float* __restrict__ ln_bias,
                                 const float* __restrict__ vln_w,
                                 bf16_t* __restrict__ x_bf,
                                 float* __restrict__ vec_f32,
                                 bf16_t* __restrict__ vec_bf) {
    __shared__ float red_mu[4];
    __shared__ float red_var[4];
    const unsigned n = blockIdx.x;
    const unsigned t = threadIdx.x;

    float xv = node_feats[n * H + t];
    float s = wave_sum32(xv);
    if ((t & 31) == 0) red_mu[t >> 5] = s;
    __syncthreads();
    float mu = (red_mu[0] + red_mu[1] + red_mu[2] + red_mu[3]) * (1.0f / 128.0f);
    float xc = xv - mu;
    float s2 = wave_sum32(xc * xc);
    if ((t & 31) == 0) red_var[t >> 5] = s2;
    __syncthreads();
    float var = (red_var[0] + red_var[1] + red_var[2] + red_var[3]) * (1.0f / 128.0f);
    float x = xc * rsqrtf(var + 1e-5f) * ln_scale[t] + ln_bias[t];
    x_bf[n * H + t] = (bf16_t)x;

    float w = vln_w[t];
#pragma unroll
    for (int l = 0; l < LDIM; ++l) {
        unsigned idx = (n * LDIM + l) * H + t;
        float vv = vector_feats[idx] * w;
        vec_f32[idx] = vv;
        vec_bf[idx]  = (bf16_t)vv;
    }
}

// ---------------------------------------------------------------------------
// WMMA GEMM: C[M,NC] = act( A[M,128](bf16) @ B[128,NC] + bias )
// B pre-packed in fragment order, register resident, reused over 4 M tiles.
// Bias folded into the WMMA C operand. NC/ACT compile-time.
// ---------------------------------------------------------------------------
template <int NC, int ACT>
__global__ void __launch_bounds__(32, 1)
gemm_bf16_wmma(const bf16_t* __restrict__ A,
               const bf16_t* __restrict__ Bp,
               const float* __restrict__ bias,
               float* __restrict__ C, int M) {
    const unsigned lane = threadIdx.x;     // 0..31
    const int mt0  = blockIdx.x * 4;       // first of 4 M tiles
    const int nt   = blockIdx.y;
    const unsigned l15 = lane & 15;
    const unsigned hi  = lane >> 4;
    const int Mtiles = M >> 4;

    // ---- B fragments: 2 contiguous 16B loads per k-chunk ----
    const bf16_t* bbase = Bp + (unsigned)nt * 2048 + lane * 16;
    v16bf bfrag[4];
#pragma unroll
    for (int kc = 0; kc < 4; ++kc) {
        bfx8 blo = *reinterpret_cast<const bfx8*>(bbase + kc * 512);
        bfx8 bhi = *reinterpret_cast<const bfx8*>(bbase + kc * 512 + 8);
        bfrag[kc] = __builtin_shufflevector(blo, bhi,
                        0, 1, 2, 3, 4, 5, 6, 7, 8, 9, 10, 11, 12, 13, 14, 15);
    }

    const unsigned col = (unsigned)nt * 16 + l15;
    const float bv = bias ? bias[col] : 0.0f;

#pragma unroll
    for (int sub = 0; sub < 4; ++sub) {
        const int mt = mt0 + sub;
        if (mt < Mtiles) {
            const bf16_t* Arow = A + (unsigned)(mt * 16 + l15) * 128 + hi * 16;
            bfx8 araw[8];
#pragma unroll
            for (int kc = 0; kc < 4; ++kc) {
                araw[2 * kc]     = *reinterpret_cast<const bfx8*>(Arow + kc * 32);
                araw[2 * kc + 1] = *reinterpret_cast<const bfx8*>(Arow + kc * 32 + 8);
            }
            v8f acc = {bv, bv, bv, bv, bv, bv, bv, bv};
#pragma unroll
            for (int kc = 0; kc < 4; ++kc) {
                v16bf a = __builtin_shufflevector(araw[2 * kc], araw[2 * kc + 1],
                              0, 1, 2, 3, 4, 5, 6, 7, 8, 9, 10, 11, 12, 13, 14, 15);
                acc = __builtin_amdgcn_wmma_f32_16x16x32_bf16(
                    false, a, false, bfrag[kc], (short)0, acc, false, false);
            }
            float* Cp = C + (unsigned)(mt * 16 + hi * 8) * NC + col;
#pragma unroll
            for (int r = 0; r < 8; ++r) {
                float val = acc[r];
                if (ACT == 1) val = silu_f(val);
                Cp[r * NC] = val;
            }
        }
    }
}

// ---------------------------------------------------------------------------
// Edge attention + vj.  One block (128 thr) per edge; head = 16 lanes.
// ---------------------------------------------------------------------------
__global__ void edge_attn_vj_kernel(const float* __restrict__ q,
                                    const float* __restrict__ k,
                                    const float* __restrict__ v,
                                    const float* __restrict__ dk,
                                    const float* __restrict__ dv,
                                    const float* __restrict__ dist,
                                    const int* __restrict__ senders,
                                    const int* __restrict__ receivers,
                                    float* __restrict__ vj,
                                    bf16_t* __restrict__ vj_bf) {
    const unsigned e = blockIdx.x;
    const unsigned t = threadIdx.x;
    const unsigned sn = (unsigned)senders[e];
    const unsigned rc = (unsigned)receivers[e];

    float p = q[rc * H + t] * k[sn * H + t] * dk[e * H + t];
    p += __shfl_xor(p, 1);
    p += __shfl_xor(p, 2);
    p += __shfl_xor(p, 4);
    p += __shfl_xor(p, 8);

    float d = dist[e];
    float cut = (d < 5.0f) ? 0.5f * (__cosf(d * 0.62831853071795864f) + 1.0f) : 0.0f;
    float attn = silu_f(p) * cut;

    float out = v[sn * H + t] * dv[e * H + t] * attn;
    vj[e * H + t]    = out;
    vj_bf[e * H + t] = (bf16_t)out;
}

// ---------------------------------------------------------------------------
// vec_dot[n,t] = sum_l vec1*vec2  (vecW = [N*8, 384])
// ---------------------------------------------------------------------------
__global__ void vec_dot_kernel(const float* __restrict__ vecW,
                               float* __restrict__ vec_dot) {
    const unsigned n = blockIdx.x;
    const unsigned t = threadIdx.x;
    float acc = 0.f;
#pragma unroll
    for (int l = 0; l < LDIM; ++l) {
        unsigned base = (n * LDIM + l) * 384;
        acc += vecW[base + t] * vecW[base + 128 + t];
    }
    vec_dot[n * H + t] = acc;
}

// ---------------------------------------------------------------------------
// Edge scatter: node_agg[r] += vj ; vec_agg[r] += vec_j*s1 + s2*d_ij
// ---------------------------------------------------------------------------
__global__ void edge_scatter_kernel(const float* __restrict__ vj,
                                    const float* __restrict__ s,
                                    const float* __restrict__ vec,
                                    const float* __restrict__ d_ij,
                                    const int* __restrict__ senders,
                                    const int* __restrict__ receivers,
                                    float* __restrict__ node_agg,
                                    float* __restrict__ vec_agg) {
    const unsigned e = blockIdx.x;
    const unsigned t = threadIdx.x;
    const unsigned sn = (unsigned)senders[e];
    const unsigned rc = (unsigned)receivers[e];

    atomic_add_f32(&node_agg[rc * H + t], vj[e * H + t]);

    float s1 = s[e * 256 + t];
    float s2 = s[e * 256 + 128 + t];
#pragma unroll
    for (int l = 0; l < LDIM; ++l) {
        float m = vec[(sn * LDIM + l) * H + t] * s1 + s2 * d_ij[e * LDIM + l];
        atomic_add_f32(&vec_agg[(rc * LDIM + l) * H + t], m);
    }
}

// ---------------------------------------------------------------------------
// Node epilogue: dx = vec_dot*o2 + o3 ; dvec = vec3*o1 + vec_agg
// ---------------------------------------------------------------------------
__global__ void node_out_kernel(const float* __restrict__ o,
                                const float* __restrict__ vec_dot,
                                const float* __restrict__ vecW,
                                const float* __restrict__ vec_agg,
                                float* __restrict__ dx,
                                float* __restrict__ dvec) {
    const unsigned n = blockIdx.x;
    const unsigned t = threadIdx.x;
    float o1 = o[n * 384 + t];
    float o2 = o[n * 384 + 128 + t];
    float o3 = o[n * 384 + 256 + t];
    dx[n * H + t] = vec_dot[n * H + t] * o2 + o3;
#pragma unroll
    for (int l = 0; l < LDIM; ++l) {
        unsigned vi = n * LDIM + l;
        dvec[vi * H + t] = vecW[vi * 384 + 256 + t] * o1 + vec_agg[vi * H + t];
    }
}

// ---------------------------------------------------------------------------
// Edge update: rejections + w_dot ; df = silu(edge@Wf) * w_dot
// ---------------------------------------------------------------------------
__global__ void edge_update_kernel(const float* __restrict__ vecWtrg,
                                   const float* __restrict__ vecWsrc,
                                   const float* __restrict__ f_silu,
                                   const float* __restrict__ d_ij,
                                   const int* __restrict__ senders,
                                   const int* __restrict__ receivers,
                                   float* __restrict__ df) {
    const unsigned e = blockIdx.x;
    const unsigned t = threadIdx.x;
    const unsigned sn = (unsigned)senders[e];
    const unsigned rc = (unsigned)receivers[e];

    float d[LDIM], w1[LDIM], w2[LDIM];
    float p1 = 0.f, p2 = 0.f;
#pragma unroll
    for (int l = 0; l < LDIM; ++l) {
        d[l]  = d_ij[e * LDIM + l];
        w1[l] = vecWtrg[(rc * LDIM + l) * H + t];
        w2[l] = vecWsrc[(sn * LDIM + l) * H + t];
        p1 += w1[l] * d[l];
        p2 += w2[l] * (-d[l]);
    }
    float wd = 0.f;
#pragma unroll
    for (int l = 0; l < LDIM; ++l) {
        float a = w1[l] - p1 * d[l];
        float b = w2[l] - p2 * (-d[l]);
        wd += a * b;
    }
    df[e * H + t] = f_silu[e * H + t] * wd;
}

// ---------------------------------------------------------------------------
// launch
// ---------------------------------------------------------------------------
extern "C" void kernel_launch(void* const* d_in, const int* in_sizes, int n_in,
                              void* d_out, int out_size, void* d_ws, size_t ws_size,
                              hipStream_t stream) {
    const int N = NNODES, E = NEDGES;

    const float* node_feats   = (const float*)d_in[0];
    const float* edge_feats   = (const float*)d_in[1];
    const float* vector_feats = (const float*)d_in[2];
    const float* distances    = (const float*)d_in[3];
    const float* d_ij         = (const float*)d_in[4];
    const int*   senders      = (const int*)  d_in[5];
    const int*   receivers    = (const int*)  d_in[6];
    const float* ln_scale     = (const float*)d_in[7];
    const float* ln_bias      = (const float*)d_in[8];
    const float* vln_w        = (const float*)d_in[9];
    const float* W_vec        = (const float*)d_in[10];
    const float* Wq = (const float*)d_in[11]; const float* bq = (const float*)d_in[12];
    const float* Wk = (const float*)d_in[13]; const float* bk = (const float*)d_in[14];
    const float* Wv = (const float*)d_in[15]; const float* bv = (const float*)d_in[16];
    const float* Wdk = (const float*)d_in[17]; const float* bdk = (const float*)d_in[18];
    const float* Wdv = (const float*)d_in[19]; const float* bdv = (const float*)d_in[20];
    const float* Ws = (const float*)d_in[21]; const float* bs = (const float*)d_in[22];
    const float* Wo = (const float*)d_in[23]; const float* bo = (const float*)d_in[24];
    const float* Wf = (const float*)d_in[25]; const float* bfb = (const float*)d_in[26];
    const float* Wsrc = (const float*)d_in[27];
    const float* Wtrg = (const float*)d_in[28];

    // ---- workspace carve ----
    char* base = (char*)d_ws;
    size_t off = 0;
    auto carve = [&](size_t bytes) -> char* {
        char* p = base + off;
        off += (bytes + 255) & ~(size_t)255;
        return p;
    };
    bf16_t* wq_p   = (bf16_t*)carve((size_t)128 * 128 * 2);
    bf16_t* wk_p   = (bf16_t*)carve((size_t)128 * 128 * 2);
    bf16_t* wv_p   = (bf16_t*)carve((size_t)128 * 128 * 2);
    bf16_t* wdk_p  = (bf16_t*)carve((size_t)128 * 128 * 2);
    bf16_t* wdv_p  = (bf16_t*)carve((size_t)128 * 128 * 2);
    bf16_t* wvec_p = (bf16_t*)carve((size_t)128 * 384 * 2);
    bf16_t* ws_p   = (bf16_t*)carve((size_t)128 * 256 * 2);
    bf16_t* wo_p   = (bf16_t*)carve((size_t)128 * 384 * 2);
    bf16_t* wf_p   = (bf16_t*)carve((size_t)128 * 128 * 2);
    bf16_t* wsrc_p = (bf16_t*)carve((size_t)128 * 128 * 2);
    bf16_t* wtrg_p = (bf16_t*)carve((size_t)128 * 128 * 2);

    bf16_t* x_bf    = (bf16_t*)carve((size_t)N * H * 2);
    bf16_t* edge_bf = (bf16_t*)carve((size_t)E * H * 2);
    float*  vec_f32 = (float*) carve((size_t)N * LDIM * H * 4);
    bf16_t* vec_bf  = (bf16_t*)carve((size_t)N * LDIM * H * 2);
    float*  qbuf    = (float*) carve((size_t)N * H * 4);
    float*  kbuf    = (float*) carve((size_t)N * H * 4);
    float*  vbuf    = (float*) carve((size_t)N * H * 4);
    float*  dkdv    = (float*) carve((size_t)E * 256 * 4);  // dk | dv, aliased as s later
    float*  dkbuf   = dkdv;
    float*  dvbuf   = dkdv + (size_t)E * H;
    float*  sbuf    = dkdv;
    float*  vecW    = (float*) carve((size_t)N * LDIM * 384 * 4);
    float*  vecWtrg = (float*) carve((size_t)N * LDIM * H * 4);
    float*  vecWsrc = (float*) carve((size_t)N * LDIM * H * 4);
    float*  vjbuf   = (float*) carve((size_t)E * H * 4);    // aliased as f_silu later
    float*  fbuf    = vjbuf;
    bf16_t* vj_bf   = (bf16_t*)carve((size_t)E * H * 2);
    float*  vecdot  = (float*) carve((size_t)N * H * 4);
    float*  nagg    = (float*) carve((size_t)N * H * 4);
    bf16_t* nagg_bf = (bf16_t*)carve((size_t)N * H * 2);
    float*  vagg    = (float*) carve((size_t)N * LDIM * H * 4);
    float*  obuf    = (float*) carve((size_t)N * 384 * 4);

    float* out      = (float*)d_out;
    float* out_dx   = out;
    float* out_df   = out + (size_t)N * H;
    float* out_dvec = out + (size_t)N * H + (size_t)E * H;

    auto cvt = [&](const float* src, bf16_t* dst, size_t n) {
        f32_to_bf16_kernel<<<(unsigned)((n + 255) / 256), 256, 0, stream>>>(
            src, dst, (unsigned)n);
    };
    auto pack = [&](const float* W, bf16_t* P, int Nc) {
        pack_weight_kernel<<<(128 * Nc + 255) / 256, 256, 0, stream>>>(W, P, Nc);
    };
    auto gemm = [&](const bf16_t* A, const bf16_t* Bp, const float* bias,
                    float* C, int M, int Nc, int act) {
        dim3 g((M / 16 + 3) / 4, Nc / 16);
        if (Nc == 128 && act == 0)
            gemm_bf16_wmma<128, 0><<<g, 32, 0, stream>>>(A, Bp, bias, C, M);
        else if (Nc == 128 && act == 1)
            gemm_bf16_wmma<128, 1><<<g, 32, 0, stream>>>(A, Bp, bias, C, M);
        else if (Nc == 256 && act == 1)
            gemm_bf16_wmma<256, 1><<<g, 32, 0, stream>>>(A, Bp, bias, C, M);
        else if (Nc == 384 && act == 0)
            gemm_bf16_wmma<384, 0><<<g, 32, 0, stream>>>(A, Bp, bias, C, M);
    };

    // weights -> packed bf16 fragments
    pack(Wq, wq_p, 128);   pack(Wk, wk_p, 128);   pack(Wv, wv_p, 128);
    pack(Wdk, wdk_p, 128); pack(Wdv, wdv_p, 128);
    pack(W_vec, wvec_p, 384);
    pack(Ws, ws_p, 256);   pack(Wo, wo_p, 384);
    pack(Wf, wf_p, 128);   pack(Wsrc, wsrc_p, 128); pack(Wtrg, wtrg_p, 128);

    // node prep + activations -> bf16
    node_prep_kernel<<<N, 128, 0, stream>>>(node_feats, vector_feats, ln_scale,
                                            ln_bias, vln_w, x_bf, vec_f32, vec_bf);
    cvt(edge_feats, edge_bf, (size_t)E * H);

    // projections (WMMA)
    gemm(x_bf, wq_p, bq, qbuf, N, 128, 0);
    gemm(x_bf, wk_p, bk, kbuf, N, 128, 0);
    gemm(x_bf, wv_p, bv, vbuf, N, 128, 0);
    gemm(edge_bf, wdk_p, bdk, dkbuf, E, 128, 1);
    gemm(edge_bf, wdv_p, bdv, dvbuf, E, 128, 1);
    gemm(vec_bf, wvec_p, nullptr, vecW, N * LDIM, 384, 0);
    gemm(vec_bf, wtrg_p, nullptr, vecWtrg, N * LDIM, 128, 0);
    gemm(vec_bf, wsrc_p, nullptr, vecWsrc, N * LDIM, 128, 0);

    vec_dot_kernel<<<N, 128, 0, stream>>>(vecW, vecdot);

    // edge attention + vj
    edge_attn_vj_kernel<<<E, 128, 0, stream>>>(qbuf, kbuf, vbuf, dkbuf, dvbuf,
                                               distances, senders, receivers,
                                               vjbuf, vj_bf);

    // s = silu(vj @ Ws + bs)   (aliases dk/dv region — stream-ordered, safe)
    gemm(vj_bf, ws_p, bs, sbuf, E, 256, 1);

    // scatter
    hipMemsetAsync(nagg, 0, (size_t)N * H * 4, stream);
    hipMemsetAsync(vagg, 0, (size_t)N * LDIM * H * 4, stream);
    edge_scatter_kernel<<<E, 128, 0, stream>>>(vjbuf, sbuf, vec_f32, d_ij,
                                               senders, receivers, nagg, vagg);

    // node output path
    cvt(nagg, nagg_bf, (size_t)N * H);
    gemm(nagg_bf, wo_p, bo, obuf, N, 384, 0);
    node_out_kernel<<<N, 128, 0, stream>>>(obuf, vecdot, vecW, vagg, out_dx, out_dvec);

    // edge feature update (f_silu aliases vj region — safe after scatter)
    gemm(edge_bf, wf_p, bfb, fbuf, E, 128, 1);
    edge_update_kernel<<<E, 128, 0, stream>>>(vecWtrg, vecWsrc, fbuf, d_ij,
                                              senders, receivers, out_df);
}
